// ChunkwiseDeltaCore_19670950216036
// MI455X (gfx1250) — matrix operand order repaired and checked by reading, not compiled, not run on hardware
//
#include <hip/hip_runtime.h>
#include <hip/hip_bf16.h>

// ---------------- problem constants ----------------
#define DIMN   1024
#define HD     64
#define NH     16
#define CHK    128
#define BS     4
#define TLEN   4096
#define NC     32            // TLEN / CHK
#define MROWS  (BS * TLEN)   // 16384
#define NMAT   5

typedef __attribute__((ext_vector_type(16))) __bf16 v16bf;
typedef __attribute__((ext_vector_type(8)))  __bf16 v8bf;
typedef __attribute__((ext_vector_type(8)))  float  v8f;

union FragU { v16bf v; v8bf h[2]; };

// A-fragment: lane holds row m, K = {g*8..g*8+7} then {16+g*8..16+g*8+7}
__device__ __forceinline__ v16bf ldfragA(const __bf16* p) {
  FragU u; u.h[0] = *(const v8bf*)p; u.h[1] = *(const v8bf*)(p + 16); return u.v;
}
// B-fragment: lane holds column n, K = {g*8..g*8+7} then {16+g*8..16+g*8+7}
__device__ __forceinline__ v16bf ldfragB(const __bf16* p) {
  FragU u; u.h[0] = *(const v8bf*)p; u.h[1] = *(const v8bf*)(p + 16); return u.v;
}
__device__ __forceinline__ v8f vzero8() {
  v8f z = {0.f,0.f,0.f,0.f,0.f,0.f,0.f,0.f}; return z;
}
__device__ __forceinline__ float softplusf(float x) {
  return (x > 20.f) ? x : log1pf(__expf(x));
}

#define WMMA_BF16(A,B,C) \
  __builtin_amdgcn_wmma_f32_16x16x32_bf16(false,(A),false,(B),(short)0,(C),false,false)

// ---------------- CDNA5 async global->LDS copy (ASYNCcnt path) ----------------
// LDS aperture maps flat addr[31:0] directly to the wave's LDS address (ISA 10.2),
// so truncating a generic pointer to a __shared__ object yields the LDS offset.
__device__ __forceinline__ uint32_t lds_off(const void* p) {
  return (uint32_t)(uintptr_t)p;
}
__device__ __forceinline__ void async_ld_b128(uint32_t lds, const void* g) {
  asm volatile("global_load_async_to_lds_b128 %0, %1, off"
               :: "v"(lds), "v"((uint64_t)(uintptr_t)g)
               : "memory");
}
#if __has_builtin(__builtin_amdgcn_s_wait_asynccnt)
#define WAIT_ASYNC0() __builtin_amdgcn_s_wait_asynccnt(0)
#define WAIT_ASYNC4() __builtin_amdgcn_s_wait_asynccnt(4)
#else
#define WAIT_ASYNC0() asm volatile("s_wait_asynccnt 0x0" ::: "memory")
#define WAIT_ASYNC4() asm volatile("s_wait_asynccnt 0x4" ::: "memory")
#endif

// ---------------- K0: fp32 -> bf16 convert ----------------
__global__ void cvt_bf16_kernel(const float* __restrict__ src,
                                __bf16* __restrict__ dst, int n) {
  int i = blockIdx.x * blockDim.x + threadIdx.x;
  if (i < n) dst[i] = (__bf16)src[i];
}

// ---------------- K1: fused 5-way projection GEMM ----------------
// out[m, n] = sum_k x[m,k] * W[n,k] (+bias, + per-matrix epilogue)
// grid: (MROWS/128, NMAT*1024/128) = (128, 40), block 256 (8 waves)
// Double-buffered LDS staged via global_load_async_to_lds_b128.
__global__ __launch_bounds__(256) void proj_gemm_kernel(
    const __bf16* __restrict__ X,     // [16384,1024] bf16
    const __bf16* __restrict__ Wall,  // [5*1024,1024] bf16 (q,k,v,beta,alpha)
    const float* __restrict__ bq, const float* __restrict__ bk,
    const float* __restrict__ bv, const float* __restrict__ bbeta,
    const float* __restrict__ balpha,
    __bf16* __restrict__ qO, __bf16* __restrict__ kO,
    __bf16* __restrict__ vO, __bf16* __restrict__ betaO,
    float* __restrict__ alphaO,       // d_out alpha_val region [B,T,H,d]
    float* __restrict__ laO)          // workspace log_alpha  [B,T,H,d]
{
  __shared__ alignas(16) __bf16 As[2 * 128 * 32];  // x tiles, [buf][m][k]
  __shared__ alignas(16) __bf16 Bs[2 * 128 * 32];  // W tiles, [buf][n][k]

  const int tid  = threadIdx.x;
  const int lane = tid & 31;
  const int w    = tid >> 5;
  const int wm   = w >> 1;            // 0..3 -> 32-row strip
  const int wn   = w & 1;             // 0..1 -> 64-col strip
  const int fm   = lane & 15;
  const int g    = lane >> 4;

  const int mBase  = blockIdx.x * 128;
  const int nBlock = blockIdx.y * 128;        // global over 5*1024

  v8f acc[2][4];
#pragma unroll
  for (int i = 0; i < 2; i++)
#pragma unroll
    for (int j = 0; j < 4; j++) acc[i][j] = vzero8();

  const int lr   = tid >> 1;            // 0..127 loader row
  const int lc   = (tid & 1) * 16;      // 0 or 16 (halves)
  const int sOff = lr * 32 + lc;

  const __bf16* gA0 = X    + (size_t)(mBase  + lr) * DIMN + lc;
  const __bf16* gB0 = Wall + (size_t)(nBlock + lr) * DIMN + lc;

  // prologue: stage k0 = 0 into buffer 0 (async, no VGPR round trip)
  async_ld_b128(lds_off(As + sOff),     gA0);
  async_ld_b128(lds_off(As + sOff + 8), gA0 + 8);
  async_ld_b128(lds_off(Bs + sOff),     gB0);
  async_ld_b128(lds_off(Bs + sOff + 8), gB0 + 8);

  int buf = 0;
  for (int k0 = 0; k0 < DIMN; k0 += 32, buf ^= 1) {
    const int nb = buf ^ 1;
    if (k0 + 32 < DIMN) {
      // issue next tile while current one is consumed
      const __bf16* gA = gA0 + k0 + 32;
      const __bf16* gB = gB0 + k0 + 32;
      async_ld_b128(lds_off(As + nb * 4096 + sOff),     gA);
      async_ld_b128(lds_off(As + nb * 4096 + sOff + 8), gA + 8);
      async_ld_b128(lds_off(Bs + nb * 4096 + sOff),     gB);
      async_ld_b128(lds_off(Bs + nb * 4096 + sOff + 8), gB + 8);
      if (k0 + 64 < DIMN) {                 // L2 prefetch one more tile out
        __builtin_prefetch(gA + 32, 0, 1);
        __builtin_prefetch(gB + 32, 0, 1);
      }
      WAIT_ASYNC4();   // oldest 4 (current buffer) complete; next 4 in flight
    } else {
      WAIT_ASYNC0();   // drain for the final tile
    }
    __syncthreads();   // all waves' pieces of the current buffer have landed

    const __bf16* Ab = As + buf * 4096;
    const __bf16* Bb = Bs + buf * 4096;
    v16bf afr[2], bfr[4];
#pragma unroll
    for (int mt = 0; mt < 2; mt++)
      afr[mt] = ldfragA(Ab + (wm * 32 + mt * 16 + fm) * 32 + g * 8);
#pragma unroll
    for (int nt = 0; nt < 4; nt++)
      bfr[nt] = ldfragB(Bb + (wn * 64 + nt * 16 + fm) * 32 + g * 8);
#pragma unroll
    for (int mt = 0; mt < 2; mt++)
#pragma unroll
      for (int nt = 0; nt < 4; nt++)
        acc[mt][nt] = WMMA_BF16(afr[mt], bfr[nt], acc[mt][nt]);

    __syncthreads();   // reads done before next iteration overwrites buf^1
  }

  // -------- epilogue --------
  const int matId = blockIdx.y >> 3;           // 0..4 (each matrix = 8 y-tiles)
  const float* bias = bq;
  if      (matId == 1) bias = bk;
  else if (matId == 2) bias = bv;
  else if (matId == 3) bias = bbeta;
  else if (matId == 4) bias = balpha;

#pragma unroll
  for (int nt = 0; nt < 4; nt++) {
    const int colg = nBlock + wn * 64 + nt * 16 + fm;   // 0..5119
    const int col  = colg & (DIMN - 1);                 // 0..1023
    const float bb = bias[col];
#pragma unroll
    for (int mt = 0; mt < 2; mt++) {
#pragma unroll
      for (int r = 0; r < 8; r++) {
        const int row = mBase + wm * 32 + mt * 16 + r + g * 8;
        const float val = acc[mt][nt][r] + bb;
        const size_t idx = (size_t)row * DIMN + col;
        if      (matId == 0) qO[idx]    = (__bf16)(val * 0.125f);   // d^-1/2
        else if (matId == 1) kO[idx]    = (__bf16)(val * 0.125f);
        else if (matId == 2) vO[idx]    = (__bf16)val;
        else if (matId == 3) betaO[idx] = (__bf16)softplusf(val);
        else {
          const float z = fminf(10.f, fmaxf(-10.f, val));
          alphaO[idx] = 1.f / (1.f + __expf(-z));   // sigmoid
          laO[idx]    = -softplusf(-z);             // log_sigmoid
        }
      }
    }
  }
}

// ---------------- K2: per-(b,n,h) chunk stats ----------------
// m = cumsum_c(mean_d log_alpha); asum = exp(m[C-1]); bmean = mean_d beta;
// vb = v * beta (bf16).  grid: B*NC*NH = 2048 blocks, 128 threads (c).
__global__ __launch_bounds__(128) void chunk_stats_kernel(
    const float*  __restrict__ la,     // [B,T,H*d]
    const __bf16* __restrict__ vB,
    const __bf16* __restrict__ betaB,
    __bf16* __restrict__ vbO,
    float* __restrict__ mO,            // [B,N,H,C]
    float* __restrict__ bmeanO,        // [B,N,H,C]
    float* __restrict__ asumO)         // [B,N,H]
{
  __shared__ float s[CHK];
  const int bid = blockIdx.x;
  const int h = bid & (NH - 1);
  const int n = (bid >> 4) & (NC - 1);
  const int b = bid >> 9;
  const int c = threadIdx.x;

  const size_t gb = ((size_t)(b * TLEN + n * CHK + c)) * DIMN + h * HD;
  float sla = 0.f, sbe = 0.f;
#pragma unroll 8
  for (int d = 0; d < HD; d++) {
    sla += la[gb + d];
    const float be = (float)betaB[gb + d];
    sbe += be;
    vbO[gb + d] = (__bf16)((float)vB[gb + d] * be);
  }
  s[c] = sla * (1.f / 64.f);
  __syncthreads();
  for (int off = 1; off < CHK; off <<= 1) {     // inclusive scan over c
    const float t = (c >= off) ? s[c - off] : 0.f;
    __syncthreads();
    s[c] += t;
    __syncthreads();
  }
  const size_t ob = ((size_t)((b * NC + n) * NH + h)) * CHK + c;
  mO[ob]     = s[c];
  bmeanO[ob] = sbe * (1.f / 64.f);
  if (c == CHK - 1) asumO[(b * NC + n) * NH + h] = __expf(s[CHK - 1]);
}

// ---------------- K3: intra-chunk attention ----------------
// attn = (q k^T) .* decay(tril);  y_intra = attn @ (v*beta)
// grid: B*NC*NH = 2048 blocks, 256 threads (8 waves, 16 rows each)
__global__ __launch_bounds__(256) void intra_attn_kernel(
    const __bf16* __restrict__ qB, const __bf16* __restrict__ kB,
    const __bf16* __restrict__ vbB,
    const float* __restrict__ mW,      // [B,N,H,C]
    float* __restrict__ yOut)          // d_out y region [B,T,D]
{
  // Phase1: QK = [q tile | k tile] (each 128x64). Phase2 reuses QK as attn[128][128].
  __shared__ alignas(16) __bf16 QK[2 * CHK * HD];   // 32 KB
  __shared__ alignas(16) __bf16 VT[HD * CHK];       // vb transposed [d][c], 16 KB
  __shared__ float mS[CHK];

  const int bid = blockIdx.x;
  const int h = bid & (NH - 1);
  const int n = (bid >> 4) & (NC - 1);
  const int b = bid >> 9;
  const int t0 = b * TLEN + n * CHK;

  const int tid  = threadIdx.x;
  const int lane = tid & 31;
  const int w    = tid >> 5;
  const int fm   = lane & 15;
  const int g    = lane >> 4;
  const int i0   = w * 16;

  // ---- stage q, k (async -> LDS) and vb^T (register transpose) ----
  const int lr = tid >> 1;           // 0..127 (c)
  const int lc = (tid & 1) * 32;     // 0 or 32 (d)
  const size_t gbase = (size_t)t0 * DIMN + h * HD;
  {
    const __bf16* gq = qB  + gbase + (size_t)lr * DIMN + lc;
    const __bf16* gk = kB  + gbase + (size_t)lr * DIMN + lc;
#pragma unroll
    for (int j = 0; j < 4; j++) {
      async_ld_b128(lds_off(QK + lr * HD + lc + j * 8),            gq + j * 8);
      async_ld_b128(lds_off(QK + CHK * HD + lr * HD + lc + j * 8), gk + j * 8);
    }
    const __bf16* gv = vbB + gbase + (size_t)lr * DIMN + lc;
#pragma unroll
    for (int j = 0; j < 32; j++) VT[(lc + j) * CHK + lr] = gv[j];
  }
  if (tid < CHK) mS[tid] = mW[((size_t)((b * NC + n) * NH + h)) * CHK + tid];
  WAIT_ASYNC0();
  __syncthreads();

  // ---- phase 1: attn = q k^T (K = 64, 2 wmma steps) ----
  v8f att[8];
#pragma unroll
  for (int jt = 0; jt < 8; jt++) att[jt] = vzero8();
#pragma unroll
  for (int kk = 0; kk < 2; kk++) {
    const v16bf a = ldfragA(QK + (i0 + fm) * HD + kk * 32 + g * 8);
#pragma unroll
    for (int jt = 0; jt < 8; jt++) {
      const v16bf bb = ldfragB(QK + CHK * HD + (jt * 16 + fm) * HD + kk * 32 + g * 8);
      att[jt] = WMMA_BF16(a, bb, att[jt]);
    }
  }
  __syncthreads();   // done reading q/k tiles; QK now becomes attn storage

  // ---- decay mask, convert to bf16, store attn in LDS ----
#pragma unroll
  for (int jt = 0; jt < 8; jt++) {
#pragma unroll
    for (int r = 0; r < 8; r++) {
      const int i = i0 + r + g * 8;
      const int j = jt * 16 + fm;
      float v = att[jt][r];
      v = (j <= i) ? v * __expf(mS[i] - mS[j]) : 0.f;
      QK[i * CHK + j] = (__bf16)v;
    }
  }
  __syncthreads();

  // ---- phase 2: y = attn @ vb (K = 128, 4 wmma steps) ----
  v8f yacc[4];
#pragma unroll
  for (int dt = 0; dt < 4; dt++) yacc[dt] = vzero8();
#pragma unroll
  for (int kk = 0; kk < 4; kk++) {
    const v16bf a = ldfragA(QK + (i0 + fm) * CHK + kk * 32 + g * 8);
#pragma unroll
    for (int dt = 0; dt < 4; dt++) {
      const v16bf bb = ldfragB(VT + (dt * 16 + fm) * CHK + kk * 32 + g * 8);
      yacc[dt] = WMMA_BF16(a, bb, yacc[dt]);
    }
  }
#pragma unroll
  for (int dt = 0; dt < 4; dt++) {
#pragma unroll
    for (int r = 0; r < 8; r++) {
      const int i = i0 + r + g * 8;
      const int d = dt * 16 + fm;
      yOut[(size_t)(t0 + i) * DIMN + h * HD + d] = yacc[dt][r];
    }
  }
}

// ---------------- K4: sequential chunk scan ----------------
// state in LDS per (b,h); mem = q@state, y += mem*curve; state = state*asum + k^T (v*bmean)
// grid: B*NH = 64 blocks, 256 threads
__global__ __launch_bounds__(256) void scan_kernel(
    const __bf16* __restrict__ qB, const __bf16* __restrict__ kB,
    const __bf16* __restrict__ vB,
    const float* __restrict__ bmeanW,  // [B,N,H,C]
    const float* __restrict__ mW,      // [B,N,H,C]
    const float* __restrict__ asumW,   // [B,N,H]
    float* __restrict__ yOut,          // updated in place
    float* __restrict__ stateOut)      // d_out state region [B,H,d,d]
{
  __shared__ float  st[HD * HD];       // state[dk][dv], 16 KB
  __shared__ __bf16 ks[CHK * HD];      // 16 KB
  __shared__ __bf16 vbm[CHK * HD];     // v * bmean, 16 KB
  const int b = blockIdx.x >> 4;
  const int h = blockIdx.x & (NH - 1);
  const int tid = threadIdx.x;
  const int lr = tid >> 1;             // 0..127 (c)
  const int lc = (tid & 1) * 32;       // 0 or 32
  const int udk = tid >> 2;            // 0..63  (update row)
  const int udv = (tid & 3) * 16;      // update col base

  for (int i = tid; i < HD * HD; i += 256) st[i] = 0.f;
  __syncthreads();

  for (int n = 0; n < NC; n++) {
    const int t0 = b * TLEN + n * CHK;
    const size_t gb = (size_t)t0 * DIMN + h * HD;
    const size_t sb = ((size_t)((b * NC + n) * NH + h)) * CHK;

    // stage k and v*bmean for this chunk
    {
      const float bm = bmeanW[sb + lr];
      const __bf16* gk = kB + gb + (size_t)lr * DIMN + lc;
      const __bf16* gv = vB + gb + (size_t)lr * DIMN + lc;
#pragma unroll
      for (int j = 0; j < 32; j++) {
        ks[lr * HD + lc + j]  = gk[j];
        vbm[lr * HD + lc + j] = (__bf16)((float)gv[j] * bm);
      }
    }
    __syncthreads();

    // mem[c, d] = sum_dk q[c,dk] * st[dk,d];  y += mem * curve[c]
    {
      float acc[32];
#pragma unroll
      for (int d = 0; d < 32; d++) acc[d] = 0.f;
      const __bf16* gq = qB + gb + (size_t)lr * DIMN;
      for (int dk = 0; dk < HD; dk++) {
        const float qv = (float)gq[dk];
#pragma unroll
        for (int d = 0; d < 32; d++) acc[d] += qv * st[dk * HD + lc + d];
      }
      const float curve = __expf(mW[sb + lr]);
      float* y = yOut + (size_t)(t0 + lr) * DIMN + h * HD + lc;
#pragma unroll
      for (int d = 0; d < 32; d++) y[d] += acc[d] * curve;
    }
    __syncthreads();   // all reads of old state done

    // state = state*asum + sum_c k[c,dk] * (v[c,dv]*bmean[c])
    {
      const float a = asumW[(b * NC + n) * NH + h];
      float acc[16];
#pragma unroll
      for (int i = 0; i < 16; i++) acc[i] = 0.f;
      for (int c = 0; c < CHK; c++) {
        const float kv = (float)ks[c * HD + udk];
#pragma unroll
        for (int i = 0; i < 16; i++) acc[i] += kv * (float)vbm[c * HD + udv + i];
      }
#pragma unroll
      for (int i = 0; i < 16; i++) {
        const int idx = udk * HD + udv + i;
        st[idx] = st[idx] * a + acc[i];
      }
    }
    __syncthreads();   // update done before next chunk's staging
  }

  // final state
  float* so = stateOut + ((size_t)(b * NH + h) * HD + udk) * HD + udv;
#pragma unroll
  for (int i = 0; i < 16; i++) so[i] = st[udk * HD + udv + i];
}

// ---------------- launch ----------------
extern "C" void kernel_launch(void* const* d_in, const int* in_sizes, int n_in,
                              void* d_out, int out_size, void* d_ws, size_t ws_size,
                              hipStream_t stream) {
  (void)in_sizes; (void)n_in; (void)out_size; (void)ws_size;
  const float* x  = (const float*)d_in[0];
  const float* Wq = (const float*)d_in[1]; const float* bq = (const float*)d_in[2];
  const float* Wk = (const float*)d_in[3]; const float* bk = (const float*)d_in[4];
  const float* Wv = (const float*)d_in[5]; const float* bv = (const float*)d_in[6];
  const float* Wb = (const float*)d_in[7]; const float* bb = (const float*)d_in[8];
  const float* Wa = (const float*)d_in[9]; const float* ba = (const float*)d_in[10];

  const size_t nX  = (size_t)MROWS * DIMN;     // 16,777,216
  const size_t nW  = (size_t)DIMN * DIMN;      //  1,048,576
  const size_t nSt = (size_t)BS * NH * HD * HD;//    262,144

  char* ws = (char*)d_ws;
  size_t off = 0;
  __bf16* xB    = (__bf16*)(ws + off); off += nX * 2;
  __bf16* Wall  = (__bf16*)(ws + off); off += NMAT * nW * 2;
  __bf16* qBuf  = (__bf16*)(ws + off); off += nX * 2;
  __bf16* kBuf  = (__bf16*)(ws + off); off += nX * 2;
  __bf16* vBuf  = (__bf16*)(ws + off); off += nX * 2;
  __bf16* beBuf = (__bf16*)(ws + off); off += nX * 2;
  __bf16* vbBuf = (__bf16*)(ws + off); off += nX * 2;
  float*  laBuf = (float*)(ws + off);  off += nX * 4;
  float*  mBuf  = (float*)(ws + off);  off += (size_t)BS * NC * NH * CHK * 4;
  float*  bmBuf = (float*)(ws + off);  off += (size_t)BS * NC * NH * CHK * 4;
  float*  asBuf = (float*)(ws + off);  off += (size_t)BS * NC * NH * 4;

  float* yOut     = (float*)d_out;          // [B,T,D]
  float* stateOut = yOut + nX;              // [B,H,d,d]
  float* alphaOut = stateOut + nSt;         // [B,T,H,d]

  // K0: bf16 conversion (x + 5 weight matrices, q/k/v/beta/alpha order)
  cvt_bf16_kernel<<<(int)((nX + 255) / 256), 256, 0, stream>>>(x,  xB, (int)nX);
  cvt_bf16_kernel<<<(int)((nW + 255) / 256), 256, 0, stream>>>(Wq, Wall + 0 * nW, (int)nW);
  cvt_bf16_kernel<<<(int)((nW + 255) / 256), 256, 0, stream>>>(Wk, Wall + 1 * nW, (int)nW);
  cvt_bf16_kernel<<<(int)((nW + 255) / 256), 256, 0, stream>>>(Wv, Wall + 2 * nW, (int)nW);
  cvt_bf16_kernel<<<(int)((nW + 255) / 256), 256, 0, stream>>>(Wb, Wall + 3 * nW, (int)nW);
  cvt_bf16_kernel<<<(int)((nW + 255) / 256), 256, 0, stream>>>(Wa, Wall + 4 * nW, (int)nW);

  // K1: fused projections (WMMA bf16, async double-buffered staging)
  dim3 g1(MROWS / 128, NMAT * DIMN / 128);
  proj_gemm_kernel<<<g1, 256, 0, stream>>>(xB, Wall, bq, bk, bv, bb, ba,
                                           qBuf, kBuf, vBuf, beBuf, alphaOut, laBuf);

  // K2: chunk stats + v*beta
  chunk_stats_kernel<<<BS * NC * NH, CHK, 0, stream>>>(laBuf, vBuf, beBuf,
                                                       vbBuf, mBuf, bmBuf, asBuf);

  // K3: intra-chunk attention (WMMA bf16) -> y_intra into d_out
  intra_attn_kernel<<<BS * NC * NH, 256, 0, stream>>>(qBuf, kBuf, vbBuf, mBuf, yOut);

  // K4: sequential scan over chunks; updates y in place, writes final state
  scan_kernel<<<BS * NH, 256, 0, stream>>>(qBuf, kBuf, vBuf, bmBuf, mBuf, asBuf,
                                           yOut, stateOut);
}
